// MLPPredictor_4724464026021
// MI455X (gfx1250) — matrix operand order repaired and checked by reading, not compiled, not run on hardware
//
#include <hip/hip_runtime.h>
#include <hip/hip_bf16.h>

typedef __attribute__((ext_vector_type(16))) __bf16 v16bf;
typedef __attribute__((ext_vector_type(8)))  float  v8f;

#define HFEAT 64
#define WAVES_PER_BLOCK 8

__global__ __launch_bounds__(256) void edge_mlp_wmma_kernel(
    const float* __restrict__ h,
    const int*   __restrict__ src,
    const int*   __restrict__ dst,
    const float* __restrict__ W1,   // [128][64] row-major (in, out)
    const float* __restrict__ b1,   // [64]
    const float* __restrict__ W2,   // [64]
    const float* __restrict__ b2,   // [1]
    float* __restrict__ score_out,  // [E]
    float* __restrict__ label_out,  // [E]
    int E, int n_tiles)
{
    __shared__ __bf16 lds_w1[128 * HFEAT];
    __shared__ float  lds_b1[HFEAT];
    __shared__ float  lds_w2[HFEAT];
    __shared__ float  lds_b2;

    const int tid = threadIdx.x;

    // ---- one-time per block: stage W1 as bf16 into LDS ----
    for (int i = tid; i < 128 * HFEAT; i += blockDim.x)
        lds_w1[i] = (__bf16)W1[i];
    if (tid < HFEAT) { lds_b1[tid] = b1[tid]; lds_w2[tid] = W2[tid]; }
    if (tid == 0)    lds_b2 = b2[0];
    __syncthreads();

    const int  lane = tid & 31;
    const int  wave = tid >> 5;
    const int  col  = lane & 15;        // N column (B,C,D) / M row (A) within half
    const bool hi   = lane >= 16;

    // A-matrix 16x32 bf16 layout: lanes 0-15 hold K 0..7 (v0-3) & 16..23 (v4-7);
    // lanes 16-31 hold K 8..15 & 24..31.
    const int ka0 = hi ? 8  : 0;
    const int ka1 = hi ? 24 : 16;
    // B-matrix 32x16 bf16 layout: same K order as A (lanes 0-15 K 0..7 & 16..23; lanes 16-31 K 8..15 & 24..31).
    const int kb0 = hi ? 16 : 0;

    // ---- hoist all W1 fragments into registers: bfrag[Kchunk][Ntile] ----
    v16bf bfrag[4][4];
#pragma unroll
    for (int c = 0; c < 4; ++c) {
#pragma unroll
        for (int n = 0; n < 4; ++n) {
            const int ncol = n * 16 + col;
#pragma unroll
            for (int i = 0; i < 16; ++i)
                bfrag[c][n][i] = lds_w1[(c * 32 + (i < 8 ? ka0 + i : ka1 + i - 8)) * HFEAT + ncol];
        }
    }

    float bias[4], w2v[4];
#pragma unroll
    for (int n = 0; n < 4; ++n) {
        bias[n] = lds_b1[n * 16 + col];
        w2v[n]  = lds_w2[n * 16 + col];
    }
    const float b2v = lds_b2;

    const v8f vzero = {0.f, 0.f, 0.f, 0.f, 0.f, 0.f, 0.f, 0.f};

    const int waveGlobal = blockIdx.x * WAVES_PER_BLOCK + wave;
    const int nWaves     = gridDim.x * WAVES_PER_BLOCK;

    for (int t = waveGlobal; t < n_tiles; t += nWaves) {
        const int base = t * 16;

        // lanes 0-15: src index of edge col; lanes 16-31: dst index of edge col
        int e = base + col; if (e >= E) e = E - 1;       // clamp tail (dup edges)
        const int nodeIdx = hi ? dst[e] : src[e];

        // node ids for this lane's A row (m = col)
        const int nsrc = __shfl(nodeIdx, col, 32);
        const int ndst = __shfl(nodeIdx, col + 16, 32);
        const float* rs = h + (size_t)nsrc * HFEAT;
        const float* rd = h + (size_t)ndst * HFEAT;

        v8f acc[4] = {vzero, vzero, vzero, vzero};

#pragma unroll
        for (int c = 0; c < 4; ++c) {
            const float* row = (c < 2) ? rs : rd;        // K 0..63 = src, 64..127 = dst
            const int kc = (c & 1) * 32;

            const float4 f0 = *(const float4*)(row + kc + ka0);
            const float4 f1 = *(const float4*)(row + kc + ka0 + 4);
            const float4 f2 = *(const float4*)(row + kc + ka1);
            const float4 f3 = *(const float4*)(row + kc + ka1 + 4);

            v16bf a;
            a[0]  = (__bf16)f0.x; a[1]  = (__bf16)f0.y;
            a[2]  = (__bf16)f0.z; a[3]  = (__bf16)f0.w;
            a[4]  = (__bf16)f1.x; a[5]  = (__bf16)f1.y;
            a[6]  = (__bf16)f1.z; a[7]  = (__bf16)f1.w;
            a[8]  = (__bf16)f2.x; a[9]  = (__bf16)f2.y;
            a[10] = (__bf16)f2.z; a[11] = (__bf16)f2.w;
            a[12] = (__bf16)f3.x; a[13] = (__bf16)f3.y;
            a[14] = (__bf16)f3.z; a[15] = (__bf16)f3.w;

#pragma unroll
            for (int n = 0; n < 4; ++n)
                acc[n] = __builtin_amdgcn_wmma_f32_16x16x32_bf16(
                    false, a, false, bfrag[c][n], (short)0, acc[n], false, false);
        }

        // ---- epilogue: bias + relu, dot with W2, reduce over columns ----
        // C layout: lane holds column (n*16+col); VGPR r holds row r (+8 if hi).
        float p[8];
#pragma unroll
        for (int r = 0; r < 8; ++r) p[r] = 0.f;
#pragma unroll
        for (int n = 0; n < 4; ++n) {
#pragma unroll
            for (int r = 0; r < 8; ++r) {
                float v = acc[n][r] + bias[n];
                v = v > 0.f ? v : 0.f;
                p[r] += v * w2v[n];
            }
        }
        // butterfly sum over the 16 lanes of each half (stays within half: masks 1,2,4,8)
#pragma unroll
        for (int m = 1; m < 16; m <<= 1) {
#pragma unroll
            for (int r = 0; r < 8; ++r)
                p[r] += __shfl_xor(p[r], m, 32);
        }

        // lanes 0 / 16 hold full sums for rows 0..7 / 8..15
        if (lane == 0 || lane == 16) {
            const int mbase = base + (hi ? 8 : 0);
#pragma unroll
            for (int r = 0; r < 8; ++r) {
                const int ei = mbase + r;
                if (ei < E) {
                    const float s = p[r] + b2v;
                    score_out[ei] = s;
                    // round(sigmoid(s)): >0 -> 1, <=0 -> 0 (round-half-even at s==0 gives 0)
                    label_out[ei] = (s > 0.f) ? 1.f : 0.f;
                }
            }
        }
    }
}

extern "C" void kernel_launch(void* const* d_in, const int* in_sizes, int n_in,
                              void* d_out, int out_size, void* d_ws, size_t ws_size,
                              hipStream_t stream) {
    const float* h   = (const float*)d_in[0];
    const int*   src = (const int*)  d_in[1];
    const int*   dst = (const int*)  d_in[2];
    const float* W1  = (const float*)d_in[3];
    const float* b1  = (const float*)d_in[4];
    const float* W2  = (const float*)d_in[5];
    const float* b2  = (const float*)d_in[6];

    const int E = in_sizes[1];
    float* score = (float*)d_out;
    float* label = score + E;

    const int n_tiles = (E + 15) / 16;
    int blocks = 1024;
    const int maxBlocks = (n_tiles + WAVES_PER_BLOCK - 1) / WAVES_PER_BLOCK;
    if (blocks > maxBlocks) blocks = maxBlocks;
    if (blocks < 1) blocks = 1;

    edge_mlp_wmma_kernel<<<blocks, 256, 0, stream>>>(
        h, src, dst, W1, b1, W2, b2, score, label, E, n_tiles);
}
